// NaturalGradientDescentVelNet_35167192220193
// MI455X (gfx1250) — matrix operand order repaired and checked by reading, not compiled, not run on hardware
//
#include <hip/hip_runtime.h>
#include <math.h>

// ---------------------------------------------------------------------------
// NaturalGradientDescentVelNet fused kernel for gfx1250 (MI455X, wave32, WMMA)
//   f16 WMMA 16x16x32, f32 accumulate. One wave = 16 batch rows, end-to-end.
//   Jacobian (jacfwd, DX=2) via stacked primal+tangent row-groups sharing B.
//   v2: branch-free activations (no EXEC-divergent libm tanh in WMMA loops),
//       2-way N-tile unroll -> 2 (or 6) independent WMMA accumulation chains
//       and A-fragment reuse across the tile pair.
// ---------------------------------------------------------------------------

typedef __attribute__((ext_vector_type(16))) _Float16 v16h;
typedef __attribute__((ext_vector_type(8)))  _Float16 v8h;
typedef __attribute__((ext_vector_type(8)))  float    v8f;

#define NPTS          131072
#define ROWS_PER_WG   64
#define WAVES_PER_WG  4
#define THREADS       128
#define BUF_ELEMS     6144                      // >= 48*128 and >= 16*320
#define SMEM_F16_BYTES (WAVES_PER_WG * 3 * BUF_ELEMS * 2)
#define SMEM_BYTES     (SMEM_F16_BYTES + WAVES_PER_WG * 144 * 4)

// ---- workspace layout (bytes), every weight block 256B aligned -------------
// f16 weights padded to [KPAD][NPAD] row-major (KPAD mult of 32, NPAD of 16)
constexpr size_t OFF_TM1W1 = 0;        // 32x112
constexpr size_t OFF_TM1W2 = 7168;     // 128x112
constexpr size_t OFF_TM1W3 = 35840;    // 128x16
constexpr size_t OFF_TM2W1 = 39936;    // 32x112
constexpr size_t OFF_TM2W2 = 47104;    // 128x112
constexpr size_t OFF_TM2W3 = 75776;    // 128x16
constexpr size_t OFF_VVW1  = 79872;    // 32x304
constexpr size_t OFF_VVW2  = 99328;    // 320x304
constexpr size_t OFF_VVW3  = 293888;   // 320x16
constexpr size_t OFF_VSW1  = 304128;   // 32x112
constexpr size_t OFF_VSW2  = 311296;   // 128x112
constexpr size_t OFF_VSW3  = 339968;   // 128x16
// f32 biases padded to NPAD
constexpr size_t OFF_TM1B1 = 344064;   // 112
constexpr size_t OFF_TM1B2 = 344512;   // 112
constexpr size_t OFF_TM1B3 = 344960;   // 16
constexpr size_t OFF_TM2B1 = 345024;
constexpr size_t OFF_TM2B2 = 345472;
constexpr size_t OFF_TM2B3 = 345920;
constexpr size_t OFF_VVB1  = 345984;   // 304
constexpr size_t OFF_VVB2  = 347200;   // 304
constexpr size_t OFF_VVB3  = 348416;   // 16
constexpr size_t OFF_VSB1  = 348480;
constexpr size_t OFF_VSB2  = 348928;
constexpr size_t OFF_VSB3  = 349376;
constexpr size_t OFF_ORIGIN = 349440;  // 2 f32: taskmap(0)

// ---------------------------------------------------------------------------
union AB { v16h v; v8h h[2]; };

__device__ __forceinline__ v8f wmma16(v16h a, v16h b, v8f c) {
  return __builtin_amdgcn_wmma_f32_16x16x32_f16(
      /*neg_a=*/false, a, /*neg_b=*/false, b,
      /*c_mod=*/(short)0, c, /*reuse_a=*/false, /*reuse_b=*/false);
}

// A fragment (16x32 f16) from LDS, rows [row0..row0+15], K-tile kt.
// ISA layout: lanes 0-15 hold K0-7 (v0-3) + K16-23 (v4-7); lanes 16-31 K8-15 + K24-31.
__device__ __forceinline__ v16h load_a(const _Float16* buf, int row0, int kt,
                                       int strideElems, int lane) {
  const int r    = row0 + (lane & 15);
  const int hsel = (lane >> 4) & 1;
  const char* p  = (const char*)(buf + (size_t)r * strideElems) + kt * 64 + hsel * 16;
  AB a;
  a.h[0] = *(const v8h*)p;          // ds_load_b128
  a.h[1] = *(const v8h*)(p + 32);   // ds_load_b128
  return a.v;
}

// B fragment (32x16 f16) from global row-major W[KPAD][npad]: lane%16 = N column,
// lanes 0-15 hold K0-7 + K16-23; lanes 16-31 K8-15 + K24-31 (matches load_a).
__device__ __forceinline__ v16h load_b(const _Float16* W, int kt, int nt,
                                       int npad, int lane) {
  const int hsel = (lane >> 4) & 1;
  const _Float16* p = W + (size_t)(kt * 32) * npad + nt * 16 + (lane & 15);
  v16h b;
  #pragma unroll
  for (int s = 0; s < 8; ++s) {
    b[s]     = p[(size_t)(8 * hsel + s) * npad];
    b[8 + s] = p[(size_t)(16 + 8 * hsel + s) * npad];
  }
  return b;
}

__device__ __forceinline__ void zero_buf(_Float16* b, int lane) {
  v8h z = {(_Float16)0,(_Float16)0,(_Float16)0,(_Float16)0,
           (_Float16)0,(_Float16)0,(_Float16)0,(_Float16)0};
  v8h* p = (v8h*)b;
  #pragma unroll
  for (int i = 0; i < BUF_ELEMS / 8 / 32; ++i) p[lane + i * 32] = z;
}

// Branch-free activation + derivative.  ACT: 0 none, 1 tanh, 2 elu,
// 3 prelu(slope), 4 leaky(0.01).  Lowers to v_exp_f32 + v_cndmask (no EXEC
// divergence inside WMMA loops).
template<int ACT>
__device__ __forceinline__ void act_fd(float u, float slope, float& h, float& d) {
  if (ACT == 1) {
    float t = fminf(fmaxf(u, -9.f), 9.f);
    float e = __expf(2.f * t);
    h = 1.f - 2.f / (e + 1.f);
    d = 1.f - h * h;
  } else if (ACT == 2) {
    float e   = __expf(fminf(u, 0.f));
    bool  pos = (u > 0.f);
    h = pos ? u : (e - 1.f);
    d = pos ? 1.f : e;
  } else if (ACT == 3) {
    bool pos = (u >= 0.f);
    h = pos ? u : slope * u;
    d = pos ? 1.f : slope;
  } else if (ACT == 4) {
    bool pos = (u >= 0.f);
    h = pos ? u : 0.01f * u;
    d = pos ? 1.f : 0.01f;
  } else {
    h = u; d = 1.f;
  }
}

// Epilogue for one N-tile: bias + activation, store h (and chain-rule-scaled
// tangents for NG==3) into the LDS output buffer.
template<int NG, int ACT>
__device__ __forceinline__ void layer_epi(const v8f* acc, int nt, _Float16* outb,
    int outStride, const float* bias, float slope, int lane)
{
  const int   n     = nt * 16 + (lane & 15);
  const float bv    = bias[n];
  const int   rbase = (lane < 16) ? 0 : 8;   // C/D layout: lanes 16-31 hold M=8..15
  #pragma unroll
  for (int r = 0; r < 8; ++r) {
    float h, d;
    act_fd<ACT>(acc[0][r] + bv, slope, h, d);
    const int row = rbase + r;
    outb[(size_t)row * outStride + n] = (_Float16)h;
    if (NG == 3) {
      outb[(size_t)(16 + row) * outStride + n] = (_Float16)(acc[1][r] * d);
      outb[(size_t)(32 + row) * outStride + n] = (_Float16)(acc[2][r] * d);
    }
  }
}

// Generic dense layer: C = act(A@W + bias).  NG row-groups (1 = primal only,
// 3 = primal + 2 tangents sharing B and scaled by act'(u_primal)).
// N-tiles processed in pairs: 2*NG independent WMMA chains, A loaded once per
// kt and reused for both tiles.
template<int NG, int ACT, int KT, int NT>
__device__ __forceinline__ void run_layer(const _Float16* inb, _Float16* outb,
    int inStride, int outStride, const _Float16* W, const float* bias,
    float slope, int lane)
{
  const int npad = NT * 16;
  const v8f zf = {0.f,0.f,0.f,0.f,0.f,0.f,0.f,0.f};
  int nt = 0;
  for (; nt + 1 < NT; nt += 2) {
    v8f acc0[NG], acc1[NG];
    #pragma unroll
    for (int g = 0; g < NG; ++g) { acc0[g] = zf; acc1[g] = zf; }
    #pragma unroll
    for (int kt = 0; kt < KT; ++kt) {
      v16h bf0 = load_b(W, kt, nt,     npad, lane);
      v16h bf1 = load_b(W, kt, nt + 1, npad, lane);
      #pragma unroll
      for (int g = 0; g < NG; ++g) {
        v16h a = load_a(inb, g * 16, kt, inStride, lane);
        acc0[g] = wmma16(a, bf0, acc0[g]);
        acc1[g] = wmma16(a, bf1, acc1[g]);
      }
    }
    layer_epi<NG, ACT>(acc0, nt,     outb, outStride, bias, slope, lane);
    layer_epi<NG, ACT>(acc1, nt + 1, outb, outStride, bias, slope, lane);
  }
  if (nt < NT) {                       // odd remainder tile
    v8f acc[NG];
    #pragma unroll
    for (int g = 0; g < NG; ++g) acc[g] = zf;
    #pragma unroll
    for (int kt = 0; kt < KT; ++kt) {
      v16h b = load_b(W, kt, nt, npad, lane);
      #pragma unroll
      for (int g = 0; g < NG; ++g) {
        v16h a = load_a(inb, g * 16, kt, inStride, lane);
        acc[g] = wmma16(a, b, acc[g]);
      }
    }
    layer_epi<NG, ACT>(acc, nt, outb, outStride, bias, slope, lane);
  }
}

// Accumulate the single N-tile (NPAD=16) of a final layer into acc[NG].
template<int NG, int KT>
__device__ __forceinline__ void accum_tile0(const _Float16* inb, int inStride,
                                            const _Float16* W, int lane, v8f* acc)
{
  #pragma unroll
  for (int kt = 0; kt < KT; ++kt) {
    v16h b = load_b(W, kt, 0, 16, lane);
    #pragma unroll
    for (int g = 0; g < NG; ++g) {
      v16h a = load_a(inb, g * 16, kt, inStride, lane);
      acc[g] = wmma16(a, b, acc[g]);
    }
  }
}

// ---------------------------------------------------------------------------
__global__ void __launch_bounds__(THREADS)
ngd_velnet_kernel(const float* __restrict__ X, const float* __restrict__ a1p,
                  const float* __restrict__ a2p, const char* __restrict__ ws,
                  float* __restrict__ out)
{
  extern __shared__ char smem[];
  const int lane = threadIdx.x & 31;
  const int wave = threadIdx.x >> 5;
  const int r0   = blockIdx.x * ROWS_PER_WG + wave * 16;   // global base row

  _Float16* b0 = (_Float16*)smem + (size_t)wave * 3 * BUF_ELEMS;
  _Float16* b1 = b0 + BUF_ELEMS;
  _Float16* b2 = b1 + BUF_ELEMS;
  float* sf    = (float*)(smem + SMEM_F16_BYTES) + wave * 144;
  float* yv    = sf;        // [16][2]   y = taskmap(x) - origin
  float* Jv    = sf + 32;   // [16][2][2] J[m][i][j]
  float* dotyv = sf + 96;   // [16][2]
  float* lsv   = sf + 128;  // [16]

  const _Float16* TM1W1 = (const _Float16*)(ws + OFF_TM1W1);
  const _Float16* TM1W2 = (const _Float16*)(ws + OFF_TM1W2);
  const _Float16* TM1W3 = (const _Float16*)(ws + OFF_TM1W3);
  const _Float16* TM2W1 = (const _Float16*)(ws + OFF_TM2W1);
  const _Float16* TM2W2 = (const _Float16*)(ws + OFF_TM2W2);
  const _Float16* TM2W3 = (const _Float16*)(ws + OFF_TM2W3);
  const _Float16* VVW1  = (const _Float16*)(ws + OFF_VVW1);
  const _Float16* VVW2  = (const _Float16*)(ws + OFF_VVW2);
  const _Float16* VVW3  = (const _Float16*)(ws + OFF_VVW3);
  const _Float16* VSW1  = (const _Float16*)(ws + OFF_VSW1);
  const _Float16* VSW2  = (const _Float16*)(ws + OFF_VSW2);
  const _Float16* VSW3  = (const _Float16*)(ws + OFF_VSW3);
  const float* TM1B1 = (const float*)(ws + OFF_TM1B1);
  const float* TM1B2 = (const float*)(ws + OFF_TM1B2);
  const float* TM1B3 = (const float*)(ws + OFF_TM1B3);
  const float* TM2B1 = (const float*)(ws + OFF_TM2B1);
  const float* TM2B2 = (const float*)(ws + OFF_TM2B2);
  const float* TM2B3 = (const float*)(ws + OFF_TM2B3);
  const float* VVB1  = (const float*)(ws + OFF_VVB1);
  const float* VVB2  = (const float*)(ws + OFF_VVB2);
  const float* VVB3  = (const float*)(ws + OFF_VVB3);
  const float* VSB1  = (const float*)(ws + OFF_VSB1);
  const float* VSB2  = (const float*)(ws + OFF_VSB2);
  const float* VSB3  = (const float*)(ws + OFF_VSB3);
  const float* ORIGIN = (const float*)(ws + OFF_ORIGIN);

  // warm L2 with the dominant 300x300 weight block (global_prefetch_b8)
  __builtin_prefetch(ws + OFF_VVW2 + (size_t)threadIdx.x * 256, 0, 1);

  const float a1 = a1p[0], a2 = a2p[0];
  const int   rbase = (lane < 16) ? 0 : 8;
  const int   ncol  = lane & 15;
  const v8f zf = {0.f,0.f,0.f,0.f,0.f,0.f,0.f,0.f};

  // ================= stage 1: taskmap nn1 (tanh MLP, residual) =============
  zero_buf(b0, lane); zero_buf(b1, lane); zero_buf(b2, lane);
  if (lane < 16) {                     // stage x + tangent unit rows (e0,e1)
    const int m = lane;
    b0[(size_t)m * 128 + 0]        = (_Float16)X[(size_t)(r0 + m) * 2 + 0];
    b0[(size_t)m * 128 + 1]        = (_Float16)X[(size_t)(r0 + m) * 2 + 1];
    b0[(size_t)(16 + m) * 128 + 0] = (_Float16)1.f;
    b0[(size_t)(32 + m) * 128 + 1] = (_Float16)1.f;
  }
  __builtin_amdgcn_wave_barrier();
  run_layer<3, 1, 1, 7>(b0, b1, 128, 128, TM1W1, TM1B1, 0.f, lane);
  __builtin_amdgcn_wave_barrier();
  run_layer<3, 1, 4, 7>(b1, b2, 128, 128, TM1W2, TM1B2, 0.f, lane);
  __builtin_amdgcn_wave_barrier();
  {  // tm1 L3: y1 = h@W3 + b3 + z  (residual = staged input rows, incl. e_j)
    v8f acc[3] = {zf, zf, zf};
    accum_tile0<3, 4>(b2, 128, TM1W3, lane, acc);
    const float bv = TM1B3[ncol];
    #pragma unroll
    for (int r = 0; r < 8; ++r) {
      const int row = rbase + r;
      float p  = acc[0][r] + bv + (float)b0[(size_t)row        * 128 + ncol];
      float t1 = acc[1][r]      + (float)b0[(size_t)(16 + row) * 128 + ncol];
      float t2 = acc[2][r]      + (float)b0[(size_t)(32 + row) * 128 + ncol];
      b0[(size_t)row        * 128 + ncol] = (_Float16)p;   // y1
      b0[(size_t)(16 + row) * 128 + ncol] = (_Float16)t1;  // dy1/dx0
      b0[(size_t)(32 + row) * 128 + ncol] = (_Float16)t2;  // dy1/dx1
    }
  }
  __builtin_amdgcn_wave_barrier();

  // ================= stage 2: taskmap nn2 (elu MLP -> softplus gate) =======
  run_layer<3, 2, 1, 7>(b0, b1, 128, 128, TM2W1, TM2B1, 0.f, lane);
  __builtin_amdgcn_wave_barrier();
  run_layer<3, 2, 4, 7>(b1, b2, 128, 128, TM2W2, TM2B2, 0.f, lane);
  __builtin_amdgcn_wave_barrier();
  {  // tm2 L3: s = softplus(g@W3+b3); y = (1+s)*y1 - origin; J[i][j]
    v8f acc[3] = {zf, zf, zf};
    accum_tile0<3, 4>(b2, 128, TM2W3, lane, acc);
    const float bv = TM2B3[ncol];
    #pragma unroll
    for (int r = 0; r < 8; ++r) {
      const int row = rbase + r;
      float spre = acc[0][r] + bv;
      // branch-free softplus / sigmoid
      float s   = fmaxf(spre, 0.f) + __logf(1.f + __expf(-fabsf(spre)));
      float sig = 1.f / (1.f + __expf(-spre));
      if (ncol < 2) {
        float y1  = (float)b0[(size_t)row        * 128 + ncol];
        float d1  = (float)b0[(size_t)(16 + row) * 128 + ncol];
        float d2  = (float)b0[(size_t)(32 + row) * 128 + ncol];
        yv[row * 2 + ncol]         = (1.f + s) * y1 - ORIGIN[ncol];
        Jv[row * 4 + ncol * 2 + 0] = sig * acc[1][r] * y1 + (1.f + s) * d1;
        Jv[row * 4 + ncol * 2 + 1] = sig * acc[2][r] * y1 + (1.f + s) * d2;
      }
    }
  }
  __builtin_amdgcn_wave_barrier();

  // ================= stage 3: vv net (2->300->300->2, PReLU) ===============
  zero_buf(b1, lane); zero_buf(b2, lane);
  if (lane < 16) {
    const int m = lane;
    b1[(size_t)m * 320 + 0] = (_Float16)yv[m * 2 + 0];
    b1[(size_t)m * 320 + 1] = (_Float16)yv[m * 2 + 1];
  }
  __builtin_amdgcn_wave_barrier();
  run_layer<1, 3, 1, 19>(b1, b2, 320, 320, VVW1, VVB1, a1, lane);
  __builtin_amdgcn_wave_barrier();
  zero_buf(b0, lane);
  __builtin_amdgcn_wave_barrier();
  run_layer<1, 3, 10, 19>(b2, b0, 320, 320, VVW2, VVB2, a2, lane);
  __builtin_amdgcn_wave_barrier();
  {  // vv L3 -> doty (16x2)
    v8f acc[1] = {zf};
    accum_tile0<1, 10>(b0, 320, VVW3, lane, acc);
    const float bv = VVB3[ncol];
    #pragma unroll
    for (int r = 0; r < 8; ++r)
      if (ncol < 2) dotyv[(rbase + r) * 2 + ncol] = acc[0][r] + bv;
  }
  __builtin_amdgcn_wave_barrier();

  // ================= stage 4: vel-scalar net (2->100->100->1, leaky) =======
  zero_buf(b1, lane); zero_buf(b2, lane);
  if (lane < 16) {
    const int m = lane;
    b1[(size_t)m * 128 + 0] = (_Float16)X[(size_t)(r0 + m) * 2 + 0];
    b1[(size_t)m * 128 + 1] = (_Float16)X[(size_t)(r0 + m) * 2 + 1];
  }
  __builtin_amdgcn_wave_barrier();
  run_layer<1, 4, 1, 7>(b1, b2, 128, 128, VSW1, VSB1, 0.f, lane);
  __builtin_amdgcn_wave_barrier();
  zero_buf(b1, lane);
  __builtin_amdgcn_wave_barrier();
  run_layer<1, 4, 4, 7>(b2, b1, 128, 128, VSW2, VSB2, 0.f, lane);
  __builtin_amdgcn_wave_barrier();
  {  // vs L3 -> scalar (16x1)
    v8f acc[1] = {zf};
    accum_tile0<1, 4>(b1, 128, VSW3, lane, acc);
    const float bv = VSB3[ncol];
    #pragma unroll
    for (int r = 0; r < 8; ++r)
      if (ncol == 0) lsv[rbase + r] = acc[0][r] + bv;
  }
  __builtin_amdgcn_wave_barrier();

  // ================= epilogue: leash, 2x2 pinv, normalize, scale ===========
  if (lane < 16) {
    const int m  = lane;
    const int gm = r0 + m;
    float y0 = yv[m * 2 + 0], y1 = yv[m * 2 + 1];
    float q0 = dotyv[m * 2 + 0], q1 = dotyv[m * 2 + 1];
    float ls = q0 * y0 + q1 * y1;
    float Vy = y0 * y0 + y1 * y1;
    float g  = ls + 1e-4f * Vy;  g = (g > 0.f) ? g : 0.f;
    float f  = g / (Vy + 1e-12f);
    float yd0 = q0 - f * y0, yd1 = q1 - f * y1;
    float nrm = sqrtf(yd0 * yd0 + yd1 * yd1);
    nrm = fmaxf(nrm, 1e-12f);
    yd0 /= nrm; yd1 /= nrm;
    float a = Jv[m * 4 + 0], b = Jv[m * 4 + 1];
    float c = Jv[m * 4 + 2], d = Jv[m * 4 + 3];
    float det = a * d - b * c;
    float i00, i01, i10, i11;
    if (fabsf(det) > 1e-20f) {               // pinv == inv when invertible
      float rd = 1.f / det;
      i00 =  d * rd; i01 = -b * rd; i10 = -c * rd; i11 = a * rd;
    } else {                                 // degenerate fallback: A^T/||A||_F^2
      float fr = a * a + b * b + c * c + d * d + 1e-12f;
      i00 = a / fr; i01 = c / fr; i10 = b / fr; i11 = d / fr;
    }
    float xh0 = i00 * yd0 + i01 * yd1;
    float xh1 = i10 * yd0 + i11 * yd1;
    float x0 = X[(size_t)gm * 2 + 0], x1 = X[(size_t)gm * 2 + 1];
    float lv = lsv[m];
    out[(size_t)gm * 2 + 0] = (__expf(lv + x0) + 1e-12f) * xh0;
    out[(size_t)gm * 2 + 1] = (__expf(lv + x1) + 1e-12f) * xh1;
  }
}

// ------------------------- prep kernels ------------------------------------
__global__ void pad_w_f16(const float* __restrict__ src, _Float16* __restrict__ dst,
                          int K, int Nsrc, int KPAD, int NPAD) {
  int i = blockIdx.x * blockDim.x + threadIdx.x;
  if (i >= KPAD * NPAD) return;
  int k = i / NPAD, n = i % NPAD;
  float v = (k < K && n < Nsrc) ? src[(size_t)k * Nsrc + n] : 0.f;
  dst[i] = (_Float16)v;
}

__global__ void pad_b_f32(const float* __restrict__ src, float* __restrict__ dst,
                          int Nsrc, int NPAD) {
  int i = blockIdx.x * blockDim.x + threadIdx.x;
  if (i >= NPAD) return;
  dst[i] = (i < Nsrc) ? src[i] : 0.f;
}

// taskmap(0) on one thread (runs once; ~21K MACs)
__global__ void origin_kernel(const float* w1, const float* bb1, const float* w2,
                              const float* bb2, const float* w3, const float* bb3,
                              const float* u1, const float* c1, const float* u2,
                              const float* c2, const float* u3, const float* c3,
                              float* originOut) {
  if (threadIdx.x != 0 || blockIdx.x != 0) return;
  float h[100], h2[100], y1[2];
  for (int j = 0; j < 100; ++j) h[j] = tanhf(bb1[j]);          // z = 0
  for (int j = 0; j < 100; ++j) {
    float u = bb2[j];
    for (int k = 0; k < 100; ++k) u += h[k] * w2[k * 100 + j];
    h2[j] = tanhf(u);
  }
  for (int n = 0; n < 2; ++n) {
    float u = bb3[n];
    for (int k = 0; k < 100; ++k) u += h2[k] * w3[k * 2 + n];
    y1[n] = u;                                                  // + z(=0)
  }
  for (int j = 0; j < 100; ++j) {
    float u = c1[j];
    for (int n = 0; n < 2; ++n) u += y1[n] * u1[n * 100 + j];
    h[j] = (u > 0.f) ? u : (__expf(u) - 1.f);
  }
  for (int j = 0; j < 100; ++j) {
    float u = c2[j];
    for (int k = 0; k < 100; ++k) u += h[k] * u2[k * 100 + j];
    h2[j] = (u > 0.f) ? u : (__expf(u) - 1.f);
  }
  for (int n = 0; n < 2; ++n) {
    float u = c3[n];
    for (int k = 0; k < 100; ++k) u += h2[k] * u3[k * 2 + n];
    float s = (u > 20.f) ? u : log1pf(__expf(u));
    originOut[n] = (1.f + s) * y1[n];
  }
  (void)w1;
}

// ---------------------------------------------------------------------------
extern "C" void kernel_launch(void* const* d_in, const int* in_sizes, int n_in,
                              void* d_out, int out_size, void* d_ws, size_t ws_size,
                              hipStream_t stream) {
  (void)in_sizes; (void)n_in; (void)out_size; (void)ws_size;
  const float* X      = (const float*)d_in[0];
  const float* tm1w1  = (const float*)d_in[1];  const float* tm1b1 = (const float*)d_in[2];
  const float* tm1w2  = (const float*)d_in[3];  const float* tm1b2 = (const float*)d_in[4];
  const float* tm1w3  = (const float*)d_in[5];  const float* tm1b3 = (const float*)d_in[6];
  const float* tm2w1  = (const float*)d_in[7];  const float* tm2b1 = (const float*)d_in[8];
  const float* tm2w2  = (const float*)d_in[9];  const float* tm2b2 = (const float*)d_in[10];
  const float* tm2w3  = (const float*)d_in[11]; const float* tm2b3 = (const float*)d_in[12];
  const float* vvw1   = (const float*)d_in[13]; const float* vvb1  = (const float*)d_in[14];
  const float* vva1   = (const float*)d_in[15];
  const float* vvw2   = (const float*)d_in[16]; const float* vvb2  = (const float*)d_in[17];
  const float* vva2   = (const float*)d_in[18];
  const float* vvw3   = (const float*)d_in[19]; const float* vvb3  = (const float*)d_in[20];
  const float* vsw1   = (const float*)d_in[21]; const float* vsb1  = (const float*)d_in[22];
  const float* vsw2   = (const float*)d_in[23]; const float* vsb2  = (const float*)d_in[24];
  const float* vsw3   = (const float*)d_in[25]; const float* vsb3  = (const float*)d_in[26];
  float* out = (float*)d_out;
  char*  ws  = (char*)d_ws;

  auto padw = [&](const float* s, size_t off, int K, int N, int KP, int NP) {
    int tot = KP * NP;
    pad_w_f16<<<(tot + 255) / 256, 256, 0, stream>>>(s, (_Float16*)(ws + off), K, N, KP, NP);
  };
  auto padb = [&](const float* s, size_t off, int N, int NP) {
    pad_b_f32<<<1, NP, 0, stream>>>(s, (float*)(ws + off), N, NP);
  };

  padw(tm1w1, OFF_TM1W1, 2, 100,  32, 112);
  padw(tm1w2, OFF_TM1W2, 100, 100, 128, 112);
  padw(tm1w3, OFF_TM1W3, 100, 2, 128, 16);
  padw(tm2w1, OFF_TM2W1, 2, 100,  32, 112);
  padw(tm2w2, OFF_TM2W2, 100, 100, 128, 112);
  padw(tm2w3, OFF_TM2W3, 100, 2, 128, 16);
  padw(vvw1,  OFF_VVW1,  2, 300,  32, 304);
  padw(vvw2,  OFF_VVW2,  300, 300, 320, 304);
  padw(vvw3,  OFF_VVW3,  300, 2, 320, 16);
  padw(vsw1,  OFF_VSW1,  2, 100,  32, 112);
  padw(vsw2,  OFF_VSW2,  100, 100, 128, 112);
  padw(vsw3,  OFF_VSW3,  100, 1, 128, 16);

  padb(tm1b1, OFF_TM1B1, 100, 112);
  padb(tm1b2, OFF_TM1B2, 100, 112);
  padb(tm1b3, OFF_TM1B3, 2, 16);
  padb(tm2b1, OFF_TM2B1, 100, 112);
  padb(tm2b2, OFF_TM2B2, 100, 112);
  padb(tm2b3, OFF_TM2B3, 2, 16);
  padb(vvb1,  OFF_VVB1,  300, 304);
  padb(vvb2,  OFF_VVB2,  300, 304);
  padb(vvb3,  OFF_VVB3,  2, 16);
  padb(vsb1,  OFF_VSB1,  100, 112);
  padb(vsb2,  OFF_VSB2,  100, 112);
  padb(vsb3,  OFF_VSB3,  1, 16);

  origin_kernel<<<1, 1, 0, stream>>>(tm1w1, tm1b1, tm1w2, tm1b2, tm1w3, tm1b3,
                                     tm2w1, tm2b1, tm2w2, tm2b2, tm2w3, tm2b3,
                                     (float*)(ws + OFF_ORIGIN));

  ngd_velnet_kernel<<<NPTS / ROWS_PER_WG, THREADS, SMEM_BYTES, stream>>>(
      X, vva1, vva2, (const char*)ws, out);
}
